// VoxelProposalLayer_19018115186952
// MI455X (gfx1250) — matrix operand order repaired and checked by reading, not compiled, not run on hardware
//
#include <hip/hip_runtime.h>
#include <hip/hip_bf16.h>

// ---------------------------------------------------------------------------
// MI455X (gfx1250) deformable-attention voxel layer.
//
// Bandwidth: full-scene transpose (2 x 128 MB) dominates (~11 us @ 23.3 TB/s);
// GEMMs (~4 GFLOP over ~10k queries) are fused into one per-query-tile kernel
// using v_wmma_f32_16x16x32_f16 with pre-packed per-lane B fragments so the
// inner loop is pure b128 loads + WMMA.
// ---------------------------------------------------------------------------

typedef __attribute__((ext_vector_type(16))) _Float16 v16h;
typedef __attribute__((ext_vector_type(8)))  _Float16 v8h;
typedef __attribute__((ext_vector_type(8)))  float    v8f;

#define SC_X 128
#define SC_Y 128
#define SC_Z 16
#define N_VOXEL (SC_X * SC_Y * SC_Z)      // 262144
#define N_PTS 100000
#define CDIM 128
#define NHEAD 8
#define NLVL 3
#define NPNT 4
#define HDIM 16
#define FFDIM 512
#define P_TOT 10080                        // 48*160 + 24*80 + 12*40
#define TQ 16                              // queries per workgroup (one WMMA M-tile)

// ---------------------------------------------------------------------------
// WMMA fragment helpers (layouts per CDNA5 ISA 7.12.2, wave32)
// ---------------------------------------------------------------------------

// A fragment: 16x32 f16 tile of row-major A. lane m = lane&15 holds row m;
// K base = kb + (lane<16 ? 0 : 8); elems 0..7 -> K=base+0..7, 8..15 -> base+16..23.
// Both halves are 16B-aligned contiguous runs -> two b128 loads.
__device__ __forceinline__ v16h load_a_frag(const _Float16* A, int lda, int kb, int lane) {
    int m = lane & 15;
    int kbase = kb + ((lane & 16) ? 8 : 0);
    const _Float16* p = A + m * lda + kbase;
    v8h lo = *(const v8h*)p;
    v8h hi = *(const v8h*)(p + 16);
    return __builtin_shufflevector(lo, hi, 0, 1, 2, 3, 4, 5, 6, 7,
                                   8, 9, 10, 11, 12, 13, 14, 15);
}

// Packed-B GEMM tile: Bp holds fragments as [nt][kk][lane][16 halfs] so each
// lane reads one contiguous 32B chunk per k-step (coalesced b128 pairs).
__device__ __forceinline__ v8f gemm_tile_p(const _Float16* A, int lda,
                                           const _Float16* Bp, int K, int nt, int lane) {
    v8f acc = {};
    int kt = K >> 5;
    const _Float16* bp = Bp + ((size_t)(nt * kt) * 32 + lane) * 16;
    for (int kk = 0; kk < kt; ++kk) {
        v16h a = load_a_frag(A, lda, kk * 32, lane);
        v16h b = *(const v16h*)bp;
        bp += 32 * 16;
        acc = __builtin_amdgcn_wmma_f32_16x16x32_f16(false, a, false, b,
                                                     (short)0, acc, false, false);
    }
    return acc;
}

// D fragment store (f32): VGPR r -> row r + (lane<16?0:8), col n0+(lane&15).
__device__ __forceinline__ void store_d_bias(float* D, int ldn, int col, int lane,
                                             v8f d, float bias) {
    int mb = (lane & 16) ? 8 : 0;
#pragma unroll
    for (int r = 0; r < 8; ++r) D[(mb + r) * ldn + col] = d[r] + bias;
}

__device__ __forceinline__ float grp8_sum(float v) {
    v += __shfl_xor(v, 1, 8);
    v += __shfl_xor(v, 2, 8);
    v += __shfl_xor(v, 4, 8);
    return v;
}

// ---------------------------------------------------------------------------
// Prep kernels
// ---------------------------------------------------------------------------

__global__ void k_mark(const int* __restrict__ pts, unsigned char* __restrict__ mask, int n) {
    int i = blockIdx.x * blockDim.x + threadIdx.x;
    if (i >= n) return;
    int x = pts[i * 3 + 0], y = pts[i * 3 + 1], z = pts[i * 3 + 2];
    if (x >= 0 && x < SC_X && y >= 0 && y < SC_Y && z >= 0 && z < SC_Z)
        mask[(x * SC_Y + y) * SC_Z + z] = 1;
}

__global__ void k_compact(const unsigned char* __restrict__ mask,
                          int* __restrict__ idx, int* __restrict__ cnt) {
    int v = blockIdx.x * blockDim.x + threadIdx.x;
    if (v < N_VOXEL && mask[v]) idx[atomicAdd(cnt, 1)] = v;
}

// Pack W (K x N row-major f32) into WMMA B-fragment order:
// Bp[((nt*(K/32)+kk)*32 + lane)*16 + e] = W[(kk*32 + ((lane&16)?8:0) + (e&7) + ((e&8)?16:0))*N + nt*16 + (lane&15)]
__global__ void k_pack_b(const float* __restrict__ W, _Float16* __restrict__ Bp,
                         int K, int N) {
    int i = blockIdx.x * blockDim.x + threadIdx.x;
    if (i >= K * N) return;
    int e    = i & 15;
    int lane = (i >> 4) & 31;
    int rest = i >> 9;
    int kt = K >> 5;
    int kk = rest % kt;
    int nt = rest / kt;
    int krow = (kk << 5) + ((lane & 16) ? 8 : 0) + (e & 7) + ((e & 8) ? 16 : 0);
    int ncol = (nt << 4) + (lane & 15);
    Bp[i] = (_Float16)W[krow * N + ncol];
}

// Gather the concatenated value matrix (P_TOT x CDIM) in f16 from (C,H,W) feats.
__global__ void k_value(const float* __restrict__ f0, const float* __restrict__ f1,
                        const float* __restrict__ f2, _Float16* __restrict__ valH) {
    int i = blockIdx.x * blockDim.x + threadIdx.x;
    if (i >= P_TOT * CDIM) return;
    int p = i >> 7, c = i & 127;
    const float* f; int hw, HW;
    if (p < 7680)      { f = f0; hw = p;        HW = 7680; }
    else if (p < 9600) { f = f1; hw = p - 7680; HW = 1920; }
    else               { f = f2; hw = p - 9600; HW = 480;  }
    valH[i] = (_Float16)f[c * HW + hw];
}

// v = value @ Wv + bv  (P_TOT x 128 @ 128 x 128), f32 result for sampling.
__global__ __launch_bounds__(128) void k_vproj(const _Float16* __restrict__ valH,
                                               const _Float16* __restrict__ WvP,
                                               const float* __restrict__ bv,
                                               float* __restrict__ vbuf) {
    int wave = threadIdx.x >> 5, lane = threadIdx.x & 31;
    int rt = blockIdx.x;                      // 630 row tiles of 16
    const _Float16* A = valH + (size_t)rt * TQ * CDIM;
    for (int t = wave; t < 8; t += 4) {
        v8f acc = gemm_tile_p(A, CDIM, WvP, CDIM, t, lane);
        int n = t * 16 + (lane & 15);
        float b = bv[n];
        int mb = (lane & 16) ? 8 : 0;
#pragma unroll
        for (int r = 0; r < 8; ++r)
            vbuf[(size_t)(rt * TQ + mb + r) * CDIM + n] = acc[r] + b;
    }
}

// LDS-tiled transpose: scene_embed (N_VOXEL x C) -> out (C x N_VOXEL).
// Global->LDS fill uses the gfx1250 async-to-LDS path (ASYNCcnt tracked).
__global__ __launch_bounds__(256) void k_transpose(const float* __restrict__ se,
                                                   float* __restrict__ out) {
    __shared__ __align__(16) float tile[32][33];
    int c0 = blockIdx.y * 32;
    long v0 = (long)blockIdx.x * 32;
    int tx = threadIdx.x & 31, ty = threadIdx.x >> 5;   // 32 x 8
#pragma unroll
    for (int r = 0; r < 32; r += 8) {
        const float* gp = se + (v0 + ty + r) * CDIM + c0 + tx;
        unsigned lds = (unsigned)(uintptr_t)&tile[ty + r][tx];
        asm volatile("global_load_async_to_lds_b32 %0, %1, off"
                     :: "v"(lds), "v"(gp) : "memory");
    }
    asm volatile("s_wait_asynccnt 0x0" ::: "memory");
    __syncthreads();
#pragma unroll
    for (int r = 0; r < 32; r += 8)
        out[(long)(c0 + ty + r) * N_VOXEL + v0 + tx] = tile[tx][ty + r];
}

// ---------------------------------------------------------------------------
// Fused deformable attention + FFN: 16 queries per workgroup, 4 waves.
// ---------------------------------------------------------------------------
__global__ __launch_bounds__(128) void k_attn(
    const float* __restrict__ se, const float* __restrict__ sp,
    const float* __restrict__ ref_pix,
    const int* __restrict__ idx, const int* __restrict__ cnt,
    const _Float16* __restrict__ WsP, const float* __restrict__ bs,
    const _Float16* __restrict__ WaP, const float* __restrict__ ba,
    const _Float16* __restrict__ WoP, const float* __restrict__ bo,
    const _Float16* __restrict__ Wf1P, const float* __restrict__ bf1,
    const _Float16* __restrict__ Wf2P, const float* __restrict__ bf2,
    const float* __restrict__ ln1_s, const float* __restrict__ ln1_b,
    const float* __restrict__ ln2_s, const float* __restrict__ ln2_b,
    const float* __restrict__ vbuf, float* __restrict__ out) {

    __shared__ __align__(16) float    qf[TQ][CDIM];   // residual q
    __shared__ __align__(32) _Float16 aF[TQ][CDIM];   // A staging (qp -> attn-out -> h1)
    __shared__ __align__(32) _Float16 rF[TQ][FFDIM];  // relu(ffn1) f16
    __shared__ __align__(16) float    sc[TQ][288];    // logits / attn / h1(160..287)

    int nq = cnt[0];
    int tile = blockIdx.x;
    if (tile * TQ >= nq) return;        // block-uniform: EXEC stays full for WMMA
    int tid = threadIdx.x;
    int wave = tid >> 5, lane = tid & 31;

    // ---- load q, qp (f16) --------------------------------------------------
    for (int j = tid; j < TQ * CDIM; j += 128) {
        int r = j >> 7, c = j & 127;
        int gq = tile * TQ + r;
        float qv = 0.f, pv = 0.f;
        if (gq < nq) {
            long vox = idx[gq];
            qv = se[vox * CDIM + c];
            pv = sp[vox * CDIM + c];
        }
        qf[r][c] = qv;
        aF[r][c] = (_Float16)(qv + pv);
    }
    __syncthreads();

    // ---- offsets + attention logits: qp @ [Ws | Wa] (18 N-tiles) -----------
    for (int t = wave; t < 18; t += 4) {
        if (t < 12) {
            v8f acc = gemm_tile_p(&aF[0][0], CDIM, WsP, CDIM, t, lane);
            int n = t * 16 + (lane & 15);
            store_d_bias(&sc[0][0], 288, n, lane, acc, bs[n]);
        } else {
            int t2 = t - 12;
            v8f acc = gemm_tile_p(&aF[0][0], CDIM, WaP, CDIM, t2, lane);
            int n = t2 * 16 + (lane & 15);
            store_d_bias(&sc[0][0], 288, 192 + n, lane, acc, ba[n]);
        }
    }
    __syncthreads();

    // ---- softmax + bilinear sampling: one thread per (query, head) ---------
    {
        int qr = tid >> 3, h = tid & 7;
        int gq = tile * TQ + qr;
        long vox = (gq < nq) ? idx[gq] : 0;
        float rx = ref_pix[vox * 2 + 0], ry = ref_pix[vox * 2 + 1];

        float lg[NLVL * NPNT], mx = -1e30f;
#pragma unroll
        for (int j = 0; j < 12; ++j) { lg[j] = sc[qr][192 + h * 12 + j]; mx = fmaxf(mx, lg[j]); }
        float ssum = 0.f;
#pragma unroll
        for (int j = 0; j < 12; ++j) { lg[j] = __expf(lg[j] - mx); ssum += lg[j]; }
        float inv = 1.f / ssum;

        float4 acc4[4];
#pragma unroll
        for (int d = 0; d < 4; ++d) acc4[d] = make_float4(0.f, 0.f, 0.f, 0.f);

        const int Hs[3] = {48, 24, 12}, Wsz[3] = {160, 80, 40}, st[3] = {0, 7680, 9600};
#pragma unroll
        for (int l = 0; l < NLVL; ++l) {
            int Hl = Hs[l], Wl = Wsz[l], s0 = st[l];
#pragma unroll
            for (int p = 0; p < NPNT; ++p) {
                float aw = lg[l * 4 + p] * inv;
                int col = ((h * NLVL + l) * NPNT + p) * 2;
                float x = rx * (float)Wl + sc[qr][col + 0] - 0.5f;
                float y = ry * (float)Hl + sc[qr][col + 1] - 0.5f;
                float x0f = floorf(x), y0f = floorf(y);
                float dx = x - x0f, dy = y - y0f;
                int x0 = (int)x0f, y0 = (int)y0f;
#pragma unroll
                for (int cy = 0; cy < 2; ++cy)
#pragma unroll
                    for (int cx = 0; cx < 2; ++cx) {
                        int xi = x0 + cx, yi = y0 + cy;
                        if (xi >= 0 && xi < Wl && yi >= 0 && yi < Hl) {
                            float w = (cx ? dx : 1.f - dx) * (cy ? dy : 1.f - dy) * aw;
                            const float4* g4 = (const float4*)(vbuf +
                                (size_t)(s0 + yi * Wl + xi) * CDIM + h * HDIM);
#pragma unroll
                            for (int d = 0; d < 4; ++d) {
                                float4 g = g4[d];
                                acc4[d].x += w * g.x; acc4[d].y += w * g.y;
                                acc4[d].z += w * g.z; acc4[d].w += w * g.w;
                            }
                        }
                    }
            }
        }
#pragma unroll
        for (int d = 0; d < 4; ++d) {
            aF[qr][h * HDIM + d * 4 + 0] = (_Float16)acc4[d].x;
            aF[qr][h * HDIM + d * 4 + 1] = (_Float16)acc4[d].y;
            aF[qr][h * HDIM + d * 4 + 2] = (_Float16)acc4[d].z;
            aF[qr][h * HDIM + d * 4 + 3] = (_Float16)acc4[d].w;
        }
    }
    __syncthreads();

    // ---- attn = out @ Wo + bo ---------------------------------------------
    for (int t = wave; t < 8; t += 4) {
        v8f acc = gemm_tile_p(&aF[0][0], CDIM, WoP, CDIM, t, lane);
        int n = t * 16 + (lane & 15);
        store_d_bias(&sc[0][0], 288, n, lane, acc, bo[n]);
    }
    __syncthreads();

    // ---- LN1(attn + q): 8 lanes per query ---------------------------------
    {
        int qr = tid >> 3, seg = tid & 7;
        float loc[16];
        float s = 0.f;
#pragma unroll
        for (int k = 0; k < 16; ++k) {
            int c = seg * 16 + k;
            float v = sc[qr][c] + qf[qr][c];
            loc[k] = v; s += v;
        }
        float m = grp8_sum(s) * (1.f / CDIM);
        float vs = 0.f;
#pragma unroll
        for (int k = 0; k < 16; ++k) { float d = loc[k] - m; vs += d * d; }
        float rs = rsqrtf(grp8_sum(vs) * (1.f / CDIM) + 1e-5f);
#pragma unroll
        for (int k = 0; k < 16; ++k) {
            int c = seg * 16 + k;
            float hv = (loc[k] - m) * rs * ln1_s[c] + ln1_b[c];
            sc[qr][160 + c] = hv;              // f32 h1 for the FFN residual
            aF[qr][c] = (_Float16)hv;          // A operand for FFN1
        }
    }
    __syncthreads();

    // ---- FFN1: relu(h1 @ Wf1 + bf1) -> f16 (32 N-tiles) --------------------
    for (int t = wave; t < 32; t += 4) {
        v8f acc = gemm_tile_p(&aF[0][0], CDIM, Wf1P, CDIM, t, lane);
        int n = t * 16 + (lane & 15);
        float b = bf1[n];
        int mb = (lane & 16) ? 8 : 0;
#pragma unroll
        for (int r = 0; r < 8; ++r) rF[mb + r][n] = (_Float16)fmaxf(acc[r] + b, 0.f);
    }
    __syncthreads();

    // ---- FFN2: (16x512) @ Wf2 + bf2 + h1 ----------------------------------
    for (int t = wave; t < 8; t += 4) {
        v8f acc = gemm_tile_p(&rF[0][0], FFDIM, Wf2P, FFDIM, t, lane);
        int n = t * 16 + (lane & 15);
        float b = bf2[n];
        int mb = (lane & 16) ? 8 : 0;
#pragma unroll
        for (int r = 0; r < 8; ++r) sc[mb + r][n] = acc[r] + b + sc[mb + r][160 + n];
    }
    __syncthreads();

    // ---- LN2: 8 lanes per query -------------------------------------------
    {
        int qr = tid >> 3, seg = tid & 7;
        float loc[16];
        float s = 0.f;
#pragma unroll
        for (int k = 0; k < 16; ++k) { loc[k] = sc[qr][seg * 16 + k]; s += loc[k]; }
        float m = grp8_sum(s) * (1.f / CDIM);
        float vs = 0.f;
#pragma unroll
        for (int k = 0; k < 16; ++k) { float d = loc[k] - m; vs += d * d; }
        float rs = rsqrtf(grp8_sum(vs) * (1.f / CDIM) + 1e-5f);
#pragma unroll
        for (int k = 0; k < 16; ++k) {
            int c = seg * 16 + k;
            qf[qr][c] = (loc[k] - m) * rs * ln2_s[c] + ln2_b[c];
        }
    }
    __syncthreads();

    // ---- scatter into transposed output -----------------------------------
    for (int j = tid; j < TQ * CDIM; j += 128) {
        int r = j >> 7, c = j & 127;
        int gq = tile * TQ + r;
        if (gq < nq) {
            long vox = idx[gq];
            out[(size_t)c * N_VOXEL + vox] = qf[r][c];
        }
    }
}

// ---------------------------------------------------------------------------
// Host launcher (graph-capture safe: memset-async + kernels only)
// ---------------------------------------------------------------------------
extern "C" void kernel_launch(void* const* d_in, const int* in_sizes, int n_in,
                              void* d_out, int out_size, void* d_ws, size_t ws_size,
                              hipStream_t stream) {
    (void)in_sizes; (void)n_in; (void)out_size; (void)ws_size;

    const float* se  = (const float*)d_in[0];
    const float* f0  = (const float*)d_in[1];
    const float* f1  = (const float*)d_in[2];
    const float* f2  = (const float*)d_in[3];
    const float* sp  = (const float*)d_in[4];
    const int*   pts = (const int*)d_in[5];
    const float* ref = (const float*)d_in[6];
    const float* Wv  = (const float*)d_in[7];
    const float* bv  = (const float*)d_in[8];
    const float* Ws  = (const float*)d_in[9];
    const float* bs  = (const float*)d_in[10];
    const float* Wa  = (const float*)d_in[11];
    const float* ba  = (const float*)d_in[12];
    const float* Wo  = (const float*)d_in[13];
    const float* bo  = (const float*)d_in[14];
    const float* l1s = (const float*)d_in[15];
    const float* l1b = (const float*)d_in[16];
    const float* Wf1 = (const float*)d_in[17];
    const float* bf1 = (const float*)d_in[18];
    const float* Wf2 = (const float*)d_in[19];
    const float* bf2 = (const float*)d_in[20];
    const float* l2s = (const float*)d_in[21];
    const float* l2b = (const float*)d_in[22];
    float* out = (float*)d_out;

    char* ws = (char*)d_ws;
    size_t o = 0;
    auto align256 = [&]() { o = (o + 255) & ~(size_t)255; };
    unsigned char* mask = (unsigned char*)(ws + o); o += N_VOXEL;            align256();
    int* cnt  = (int*)(ws + o);                     o += 256;                align256();
    int* idx  = (int*)(ws + o);                     o += (size_t)N_VOXEL * 4; align256();
    _Float16* valH = (_Float16*)(ws + o);           o += (size_t)P_TOT * CDIM * 2; align256();
    float* vbuf    = (float*)(ws + o);              o += (size_t)P_TOT * CDIM * 4; align256();
    _Float16* WvP  = (_Float16*)(ws + o);           o += 128 * 128 * 2;      align256();
    _Float16* WsP  = (_Float16*)(ws + o);           o += 128 * 192 * 2;      align256();
    _Float16* WaP  = (_Float16*)(ws + o);           o += 128 * 96 * 2;       align256();
    _Float16* WoP  = (_Float16*)(ws + o);           o += 128 * 128 * 2;      align256();
    _Float16* Wf1P = (_Float16*)(ws + o);           o += 128 * 512 * 2;      align256();
    _Float16* Wf2P = (_Float16*)(ws + o);           o += 512 * 128 * 2;      align256();

    hipMemsetAsync(mask, 0, N_VOXEL, stream);
    hipMemsetAsync(cnt, 0, sizeof(int), stream);

    k_mark<<<(N_PTS + 255) / 256, 256, 0, stream>>>(pts, mask, N_PTS);
    k_compact<<<N_VOXEL / 256, 256, 0, stream>>>(mask, idx, cnt);

    k_pack_b<<<64, 256, 0, stream>>>(Wv, WvP, 128, 128);
    k_pack_b<<<96, 256, 0, stream>>>(Ws, WsP, 128, 192);
    k_pack_b<<<48, 256, 0, stream>>>(Wa, WaP, 128, 96);
    k_pack_b<<<64, 256, 0, stream>>>(Wo, WoP, 128, 128);
    k_pack_b<<<256, 256, 0, stream>>>(Wf1, Wf1P, 128, 512);
    k_pack_b<<<256, 256, 0, stream>>>(Wf2, Wf2P, 512, 128);
    k_value<<<(P_TOT * CDIM + 255) / 256, 256, 0, stream>>>(f0, f1, f2, valH);

    k_vproj<<<P_TOT / TQ, 128, 0, stream>>>(valH, WvP, bv, vbuf);

    k_transpose<<<dim3(N_VOXEL / 32, CDIM / 32), 256, 0, stream>>>(se, out);

    k_attn<<<N_VOXEL / TQ, 128, 0, stream>>>(se, sp, ref, idx, cnt,
                                             WsP, bs, WaP, ba, WoP, bo,
                                             Wf1P, bf1, Wf2P, bf2,
                                             l1s, l1b, l2s, l2b, vbuf, out);
}
